// GRNLayer_71167608095134
// MI455X (gfx1250) — matrix operand order repaired and checked by reading, not compiled, not run on hardware
//
#include <hip/hip_runtime.h>
#include <math.h>

// Problem dims (from reference): T=2048, B=64, I=256, H=256, 4H=1024
#define TSTEPS 2048
#define BATCH  64
#define INDIM  256
#define HID    256
#define GDIM   1024           // 4*H
#define ROWS   (TSTEPS*BATCH) // 131072 rows of x / xproj
#define NWG_SCAN 16           // persistent workgroups in the scan (fits one SE / cluster)

typedef __attribute__((ext_vector_type(16))) __bf16 v16bf;
typedef __attribute__((ext_vector_type(8)))  float  v8f;
typedef __attribute__((ext_vector_type(4)))  unsigned int v4u;
typedef __attribute__((ext_vector_type(8)))  int v8i;
typedef __attribute__((ext_vector_type(4)))  int v4i;

// ---------------- workspace layout (bytes) ----------------
#define OFF_BAR   0ull                            // 1 uint barrier counter
#define OFF_BSUM  1024ull                         // 1024 f32 = b_i + b_h
#define OFF_HBF   8192ull                         // h state, bf16 [64,256]    (32 KB)
#define OFF_C     65536ull                        // c state, f32  [64,256]    (64 KB)
#define OFF_WIB   131072ull                       // W_i bf16 [1024,256]       (512 KB)
#define OFF_WHB   (OFF_WIB + 524288ull)           // W_h bf16 [1024,256]       (512 KB)
#define OFF_XBF   (OFF_WHB + 524288ull)           // x bf16 [T*B,256]          (64 MB)
#define OFF_XPROJ (OFF_XBF + 67108864ull)         // xproj f32 [T*B,1024]      (512 MB)

__device__ __forceinline__ unsigned short f2bf(float f) {
  unsigned u = __float_as_uint(f);
  u += 0x7FFFu + ((u >> 16) & 1u);   // round-to-nearest-even
  return (unsigned short)(u >> 16);
}
__device__ __forceinline__ unsigned pack2(float a, float b) {
  return (unsigned)f2bf(a) | ((unsigned)f2bf(b) << 16);
}
__device__ __forceinline__ float sigmoidf_(float x) {
  return 1.0f / (1.0f + __expf(-x));
}

union ABFrag { uint4 q[2]; unsigned u[8]; v16bf v; };

// ============== kernel 1: prep (bf16 weights, bias sum, state init, barrier reset) ==============
__global__ __launch_bounds__(256) void prep_kernel(
    const float* __restrict__ Wi, const float* __restrict__ Wh,
    const float* __restrict__ bi, const float* __restrict__ bh,
    const float* __restrict__ h0, const float* __restrict__ c0,
    unsigned char* __restrict__ ws) {
  unsigned* bar = (unsigned*)(ws + OFF_BAR);
  float* bsum = (float*)(ws + OFF_BSUM);
  unsigned short* wib = (unsigned short*)(ws + OFF_WIB);
  unsigned short* whb = (unsigned short*)(ws + OFF_WHB);
  unsigned short* hbf = (unsigned short*)(ws + OFF_HBF);
  float* cst = (float*)(ws + OFF_C);

  int tid = blockIdx.x * blockDim.x + threadIdx.x;
  int nthr = gridDim.x * blockDim.x;
  if (tid == 0) *bar = 0u;                              // re-armed every launch (graph-replay safe)
  for (int i = tid; i < GDIM; i += nthr) bsum[i] = bi[i] + bh[i];
  for (int i = tid; i < GDIM * INDIM; i += nthr) { wib[i] = f2bf(Wi[i]); whb[i] = f2bf(Wh[i]); }
  for (int i = tid; i < BATCH * HID; i += nthr) { hbf[i] = f2bf(h0[i]); cst[i] = c0[i]; }
}

// ============== kernel 1b: stream-convert x fp32 -> bf16 (HBM-bandwidth bound) ==============
__global__ __launch_bounds__(256) void convert_x_kernel(
    const float* __restrict__ x, unsigned char* __restrict__ ws) {
  uint2* xbf = (uint2*)(ws + OFF_XBF);             // 8 bytes = 4 bf16
  const float4* xv = (const float4*)x;
  size_t n4 = (size_t)ROWS * INDIM / 4;            // 8,388,608 float4s
  size_t tid = (size_t)blockIdx.x * blockDim.x + threadIdx.x;
  size_t nthr = (size_t)gridDim.x * blockDim.x;
  for (size_t i = tid; i < n4; i += nthr) {
    float4 f = xv[i];
    uint2 o;
    o.x = pack2(f.x, f.y);
    o.y = pack2(f.z, f.w);
    xbf[i] = o;
  }
}

// ============== kernel 2: xproj = x @ W_i^T + (b_i+b_h), fp32 out, bf16 WMMA ==============
// One wave computes a 16(M) x 64(N) strip: 4 accumulator tiles, K=256.
// Inner loop is pure b128 loads + v_wmma; all 4 B fragments issued as one clause.
__global__ __launch_bounds__(256) void xproj_gemm_kernel(unsigned char* __restrict__ ws) {
  const unsigned short* xbf = (const unsigned short*)(ws + OFF_XBF);
  const unsigned short* wib = (const unsigned short*)(ws + OFF_WIB);
  const float* bsum = (const float*)(ws + OFF_BSUM);
  float* xproj = (float*)(ws + OFF_XPROJ);

  int wid  = (blockIdx.x * blockDim.x + threadIdx.x) >> 5;  // 131072 waves total
  int lane = threadIdx.x & 31;
  int mtile = wid >> 4;          // 8192 M-tiles
  int strip = wid & 15;          // 16 N-strips of 64
  int m0 = mtile * 16;
  int n0 = strip * 64;
  int half = lane >> 4;

  v8f acc[4];
  v8f zer = {0.f,0.f,0.f,0.f,0.f,0.f,0.f,0.f};
  #pragma unroll
  for (int nt = 0; nt < 4; ++nt) acc[nt] = zer;

  const unsigned short* arow = xbf + (size_t)(m0 + (lane & 15)) * INDIM;
  const unsigned short* brow = wib + (size_t)(n0 + (lane & 15)) * INDIM + 8 * half;

  #pragma unroll
  for (int kk = 0; kk < 8; ++kk) {
    int k0 = kk * 32;
    // A fragment (16x32 bf16): lanes 0-15 hold K{0..7,16..23}, lanes 16-31 K{8..15,24..31}
    ABFrag a;
    a.q[0] = *(const uint4*)(arow + k0 + 8 * half);
    a.q[1] = *(const uint4*)(arow + k0 + 16 + 8 * half);
    // All 4 B fragments up-front so the loads batch into one clause.
    ABFrag b[4];
    #pragma unroll
    for (int nt = 0; nt < 4; ++nt) {
      const uint4* bp = (const uint4*)(brow + (size_t)(nt * 16) * INDIM + k0);
      b[nt].q[0] = bp[0]; b[nt].q[1] = bp[2];
    }
    #pragma unroll
    for (int nt = 0; nt < 4; ++nt)
      acc[nt] = __builtin_amdgcn_wmma_f32_16x16x32_bf16(
          false, a.v, false, b[nt].v, (short)0, acc[nt], false, false);
  }

  #pragma unroll
  for (int nt = 0; nt < 4; ++nt) {
    #pragma unroll
    for (int r = 0; r < 8; ++r) {
      int row = m0 + r + 8 * half;            // C/D layout: M = r + 8*(lane>>4)
      int col = n0 + nt * 16 + (lane & 15);   // N = lane&15
      xproj[(size_t)row * GDIM + col] = acc[nt][r] + bsum[col];
    }
  }
}

// ============== grid barrier for the scan ==============
__device__ __forceinline__ void grid_barrier(unsigned* bar, unsigned target) {
  __syncthreads();
  if (threadIdx.x == 0) {
    __builtin_amdgcn_fence(__ATOMIC_RELEASE, "agent");      // publish h/c stores (global_wb)
    __atomic_fetch_add(bar, 1u, __ATOMIC_RELAXED);
    while (__atomic_load_n(bar, __ATOMIC_RELAXED) < target)
      __builtin_amdgcn_s_sleep(1);
    __builtin_amdgcn_fence(__ATOMIC_ACQUIRE, "agent");      // see other WGs' h (global_inv)
  }
  // Architected NOP when not cluster-dispatched; HW fast-path if these 16 WGs
  // are ever launched as one cluster (they fit a single SE).
  __builtin_amdgcn_s_cluster_barrier();
  __syncthreads();
}

// ============== kernel 3: persistent sequential scan ==============
// 16 WGs x 256 threads. WG b owns hidden units [16b, 16b+16): gate columns
// {j0..j0+15} of each of the 4 gates. W_h slice (32 KB bf16) is DMA'd into LDS
// by the Tensor Data Mover once and stays resident for all 2048 steps.
// Wave w: M-tile = w&3 (batch rows), gates = {2*(w>>2), +1}.
__global__ __launch_bounds__(256, 1) void scan_kernel(
    unsigned char* __restrict__ ws, float* __restrict__ out) {
  // TDM writes the 3-D tile linearly: [gate][row(16)][k(256)] bf16, 32 KB.
  __shared__ __align__(16) unsigned short whl[4][16][256];
  __shared__ __align__(16) float gl[4][BATCH][16];             // sigmoid gates, 16 KB

  unsigned* bar = (unsigned*)(ws + OFF_BAR);
  unsigned short* hbf = (unsigned short*)(ws + OFF_HBF);
  float* cst = (float*)(ws + OFF_C);
  const unsigned short* whb = (const unsigned short*)(ws + OFF_WHB);
  const float* xproj = (const float*)(ws + OFF_XPROJ);

  const int j0 = blockIdx.x * 16;      // hidden-unit slice
  const int w = threadIdx.x >> 5;
  const int lane = threadIdx.x & 31;
  const int half = lane >> 4;
  const int mt = w & 3, m0 = mt * 16;
  const int gp = (w >> 2) * 2;         // this wave's two gates

  // ---- TDM: DMA the W_h slice (4 x 16 x 256 bf16 tile) into LDS ----
  // D# per CDNA5 ISA ch.8: dims: x=k (256), y=row-in-gate (stride 256),
  // z=gate (stride 65536). data_size=2B. Tile = 256 x 16 x 4.
  if (w == 0) {
    unsigned lds_addr = (unsigned)(uintptr_t)(&whl[0][0][0]);   // flat addr[31:0] == LDS offset
    unsigned long long gaddr =
        (unsigned long long)(uintptr_t)(whb + (size_t)j0 * HID); // tile start: gate 0, row j0
    v4u g0;
    g0[0] = 1u;                                          // count=1 (valid user D#)
    g0[1] = lds_addr;                                    // lds_addr [63:32]
    g0[2] = (unsigned)(gaddr & 0xFFFFFFFFull);           // global_addr[31:0]
    g0[3] = (unsigned)((gaddr >> 32) & 0x01FFFFFFull)    // global_addr[56:32]
          | (2u << 30);                                  // type=2 ("image")
    v8i g1;
    g1[0] = (int)(1u << 16);          // wg_mask=0 (not clustered), data_size=1 -> 2 bytes
    g1[1] = (int)(256u << 16);        // tensor_dim0[15:0]=256  (bits 63:48)
    g1[2] = (int)(256u << 16);        // tensor_dim1[15:0]=256  (bits 95:80)
    g1[3] = (int)(256u << 16);        // tile_dim0=256          (bits 127:112)
    g1[4] = (int)((4u << 16) | 16u);  // tile_dim1=16, tile_dim2=4
    g1[5] = 256;                      // tensor_dim0_stride[31:0] = 256 (row stride)
    g1[6] = 0;                        // dim0_stride[47:32]=0, dim1_stride[15:0]=0
    g1[7] = 1;                        // dim1_stride[47:16]: 65536>>16 = 1 (gate stride)
    v4i g2;
    g2[0] = 4;                        // tensor_dim2 = 4 gates
    g2[1] = 0; g2[2] = 0; g2[3] = 0;  // no dim3 / no iterate
    v4i g3 = {0, 0, 0, 0};
    v8i g4 = {0, 0, 0, 0, 0, 0, 0, 0};   // extra group of the 6-arg builtin: zero-filled
    __builtin_amdgcn_tensor_load_to_lds(g0, g1, g2, g3, g4, 0);
    __builtin_amdgcn_s_wait_tensorcnt(0);
  }
  __syncthreads();

  const size_t OUT_HT = (size_t)TSTEPS * BATCH * HID;

  for (int t = 0; t < TSTEPS; ++t) {
    const float* xp = xproj + (size_t)t * (BATCH * GDIM);

    // prefetch next step's xproj rows for this wave's tiles
    if (t + 1 < TSTEPS) {
      const float* xn = xproj + (size_t)(t + 1) * (BATCH * GDIM)
                      + (size_t)(m0 + (lane & 15)) * GDIM + gp * HID + j0;
      __builtin_prefetch(xn, 0, 1);
    }

    // ---- gates partial = h @ W_h^T via WMMA ----
    v8f acc0 = {0.f,0.f,0.f,0.f,0.f,0.f,0.f,0.f};
    v8f acc1 = acc0;
    const unsigned short* hrow = hbf + (size_t)(m0 + (lane & 15)) * HID;
    // B fragment base: row (g*16 + n) of the LDS tile, k-offset 8*half
    const unsigned short* b0row = &whl[gp][lane & 15][8 * half];
    const unsigned short* b1row = &whl[gp + 1][lane & 15][8 * half];
    #pragma unroll
    for (int kk = 0; kk < 8; ++kk) {
      int k0 = kk * 32;
      ABFrag a;
      a.q[0] = *(const uint4*)(hrow + k0 + 8 * half);        // K {0..7}+8*half
      a.q[1] = *(const uint4*)(hrow + k0 + 16 + 8 * half);   // K {16..23}+8*half
      ABFrag b0, b1;
      b0.q[0] = *(const uint4*)(b0row + k0);
      b0.q[1] = *(const uint4*)(b0row + k0 + 16);
      b1.q[0] = *(const uint4*)(b1row + k0);
      b1.q[1] = *(const uint4*)(b1row + k0 + 16);
      acc0 = __builtin_amdgcn_wmma_f32_16x16x32_bf16(false, a.v, false, b0.v,
                                                     (short)0, acc0, false, false);
      acc1 = __builtin_amdgcn_wmma_f32_16x16x32_bf16(false, a.v, false, b1.v,
                                                     (short)0, acc1, false, false);
    }

    // ---- + xproj, sigmoid, stash gate tiles in LDS ----
    #pragma unroll
    for (int r = 0; r < 8; ++r) {
      int row = m0 + r + 8 * half;
      int n = lane & 15;
      float g0 = acc0[r] + xp[(size_t)row * GDIM + (gp * HID + j0 + n)];
      float g1 = acc1[r] + xp[(size_t)row * GDIM + ((gp + 1) * HID + j0 + n)];
      gl[gp][row][n]     = sigmoidf_(g0);
      gl[gp + 1][row][n] = sigmoidf_(g1);
    }
    __syncthreads();

    // ---- elementwise state update for this WG's 64x16 slice ----
    #pragma unroll
    for (int e = 0; e < 4; ++e) {
      int idx = threadIdx.x + e * 256;       // 1024 elements
      int row = idx >> 4, col = idx & 15;
      float ig = gl[0][row][col];            // gate order: in, out, z, forget
      float og = gl[1][row][col];
      float zg = gl[2][row][col];
      float fg = gl[3][row][col];
      int ci = row * HID + j0 + col;
      float cn = cst[ci] * fg + zg - ig;
      cst[ci] = cn;
      float hn = sigmoidf_(cn) - og;
      out[(size_t)t * (BATCH * HID) + ci] = hn;
      hbf[ci] = f2bf(hn);
      if (t == TSTEPS - 1) {
        out[OUT_HT + ci] = hn;                       // h_T
        out[OUT_HT + (size_t)BATCH * HID + ci] = cn; // c_T
      }
    }

    grid_barrier(bar, (unsigned)(NWG_SCAN * (t + 1)));
  }
}

// ============== host launcher ==============
extern "C" void kernel_launch(void* const* d_in, const int* in_sizes, int n_in,
                              void* d_out, int out_size, void* d_ws, size_t ws_size,
                              hipStream_t stream) {
  const float* x  = (const float*)d_in[0];
  const float* h0 = (const float*)d_in[1];
  const float* c0 = (const float*)d_in[2];
  const float* Wi = (const float*)d_in[3];
  const float* Wh = (const float*)d_in[4];
  const float* bi = (const float*)d_in[5];
  const float* bh = (const float*)d_in[6];
  float* out = (float*)d_out;
  unsigned char* ws = (unsigned char*)d_ws;
  (void)in_sizes; (void)n_in; (void)out_size; (void)ws_size;

  prep_kernel<<<256, 256, 0, stream>>>(Wi, Wh, bi, bh, h0, c0, ws);
  convert_x_kernel<<<8192, 256, 0, stream>>>(x, ws);
  // 131072 wave-tiles (8192 M-tiles x 16 N-strips), 8 waves/WG -> 16384 WGs
  xproj_gemm_kernel<<<16384, 256, 0, stream>>>(ws);
  scan_kernel<<<NWG_SCAN, 256, 0, stream>>>(ws, out);
}
